// LocalAttention_77635828843065
// MI455X (gfx1250) — matrix operand order repaired and checked by reading, not compiled, not run on hardware
//
#include <hip/hip_runtime.h>
#include <hip/hip_bf16.h>

// CDNA5 / gfx1250 wave32 WMMA types
typedef __bf16       v16bf __attribute__((ext_vector_type(16)));
typedef __bf16       v8bf  __attribute__((ext_vector_type(8)));
typedef float        v8f   __attribute__((ext_vector_type(8)));
typedef unsigned int v4u   __attribute__((ext_vector_type(4)));
typedef unsigned int v8u   __attribute__((ext_vector_type(8)));

#define HN     8
#define DMODEL 512
#define DH     64
#define BB     8
#define LL     4096
#define LSEG   128
#define WIN    192
#define W2     32
#define NSEG   32

__device__ __forceinline__ unsigned rfl(unsigned x) {
  return (unsigned)__builtin_amdgcn_readfirstlane((int)x);
}
// two contiguous 8-element (16B) loads -> one 16-element fragment
__device__ __forceinline__ v16bf ld16pair(const __bf16* p0, const __bf16* p1) {
  v8bf lo = *(const v8bf*)p0;
  v8bf hi = *(const v8bf*)p1;
  return __builtin_shufflevector(lo, hi, 0, 1, 2, 3, 4, 5, 6, 7,
                                 8, 9, 10, 11, 12, 13, 14, 15);
}

// ============================================================
// Tensor Data Mover: 2D tile (rows x cols of 2-byte elems) global -> LDS.
// D# per CDNA5 ISA. Gate to one wave; wait TENSORcnt.
// ============================================================
__device__ __forceinline__ void tdm_load_2d(unsigned lds_off, const void* gptr,
                                            unsigned rows, unsigned cols,
                                            unsigned row_stride_elems) {
  const unsigned long long ga = (unsigned long long)gptr;
  v4u g0;
  g0[0] = rfl(1u);                                   // count=1, user mode
  g0[1] = rfl(lds_off);                              // lds_addr (bytes)
  g0[2] = rfl((unsigned)ga);                         // global_addr[31:0]
  g0[3] = rfl(((unsigned)(ga >> 32) & 0x01FFFFFFu) | 0x80000000u); // [56:32] | type=2
  v8u g1;
  g1[0] = rfl(1u << 16);                             // data_size=1 (2 bytes)
  g1[1] = rfl((cols & 0xFFFFu) << 16);               // tensor_dim0 lo16 @63:48
  g1[2] = rfl((cols >> 16) | ((rows & 0xFFFFu) << 16));
  g1[3] = rfl((rows >> 16) | (cols << 16));          // | tile_dim0
  g1[4] = rfl(rows & 0xFFFFu);                       // tile_dim1, tile_dim2=0
  g1[5] = rfl(row_stride_elems);                     // tensor_dim0_stride[31:0]
  g1[6] = 0;
  g1[7] = 0;
  v4u gz; gz[0] = 0; gz[1] = 0; gz[2] = 0; gz[3] = 0;
  asm volatile("tensor_load_to_lds %0, %1, %2, %3"
               :: "s"(g0), "s"(g1), "s"(gz), "s"(gz)
               : "memory");
}

// ============================================================
// Projection GEMM: Y = W * X + bias.
// XTRANS=false: X f32 layout (b, 512, L).  XTRANS=true: X bf16 (b, L, 512).
// OUTMODE: 0 = bf16 (b,512,L), 1 = bf16 transposed (b,L,512), 2 = f32 (b,512,L).
// ============================================================
template<bool XTRANS, int OUTMODE>
__global__ __launch_bounds__(256) void
proj_kernel(const float* __restrict__ W,
            const float* __restrict__ bias,
            const void* __restrict__ Xv,
            void* __restrict__ Yv)
{
  const int lane = threadIdx.x & 31;
  const int wave = threadIdx.x >> 5;
  const int hl   = lane >> 4;
  const int q15  = lane & 15;
  const int b    = blockIdx.y;
  const int l0   = blockIdx.x * 16;
  const int ob   = wave * 64;                        // 4 o-tiles per wave
  const float* Xf  = (const float*)Xv;
  const __bf16* Xb = (const __bf16*)Xv;

  v8f acc[4];
  #pragma unroll
  for (int i = 0; i < 4; ++i) acc[i] = (v8f){};

  for (int d0 = 0; d0 < DMODEL; d0 += 32) {
    // B fragment: B[k=d][n=l]
    v16bf bb;
    if (XTRANS) {
      const __bf16* xr = Xb + ((size_t)b * LL + (l0 + q15)) * DMODEL + d0 + hl * 16;
      bb = ld16pair(xr, xr + 8);
    } else {
      #pragma unroll
      for (int e = 0; e < 16; ++e)
        bb[e] = (__bf16)Xf[(size_t)b * DMODEL * LL +
                           (size_t)(d0 + hl * 16 + e) * LL + (l0 + q15)];
    }
    #pragma unroll
    for (int i = 0; i < 4; ++i) {
      const float* wr = W + (size_t)(ob + i * 16 + q15) * DMODEL + d0 + hl * 16;
      const float4 w0 = ((const float4*)wr)[0];
      const float4 w1 = ((const float4*)wr)[1];
      const float4 w2 = ((const float4*)(wr + 8))[0];
      const float4 w3 = ((const float4*)(wr + 8))[1];
      v16bf a;
      a[0]  = (__bf16)w0.x; a[1]  = (__bf16)w0.y; a[2]  = (__bf16)w0.z; a[3]  = (__bf16)w0.w;
      a[4]  = (__bf16)w1.x; a[5]  = (__bf16)w1.y; a[6]  = (__bf16)w1.z; a[7]  = (__bf16)w1.w;
      a[8]  = (__bf16)w2.x; a[9]  = (__bf16)w2.y; a[10] = (__bf16)w2.z; a[11] = (__bf16)w2.w;
      a[12] = (__bf16)w3.x; a[13] = (__bf16)w3.y; a[14] = (__bf16)w3.z; a[15] = (__bf16)w3.w;
      acc[i] = __builtin_amdgcn_wmma_f32_16x16x32_bf16(false, a, false, bb,
                                                       (short)0, acc[i], false, false);
    }
  }

  if constexpr (OUTMODE == 1) {
    // transpose 16(l) x 64(o) tile through a per-wave LDS strip, then
    // per-lane vectorized row stores to Y[b][l][o] (pitch 72 keeps 16B align)
    __shared__ __bf16 tstrip[8 * 16 * 72];
    __bf16* ts = tstrip + wave * 16 * 72;
    #pragma unroll
    for (int i = 0; i < 4; ++i)
      #pragma unroll
      for (int r = 0; r < 8; ++r) {
        const int oo = i * 16 + 8 * hl + r;
        ts[q15 * 72 + oo] = (__bf16)(acc[i][r] + bias[ob + oo]);
      }
    __bf16* dst = (__bf16*)Yv + ((size_t)b * LL + (l0 + q15)) * DMODEL + ob;
    const __bf16* src = ts + q15 * 72;
    #pragma unroll
    for (int c = 0; c < 8; ++c)
      *(v8bf*)(dst + c * 8) = *(const v8bf*)(src + c * 8);
  } else {
    #pragma unroll
    for (int i = 0; i < 4; ++i)
      #pragma unroll
      for (int r = 0; r < 8; ++r) {
        const int o = ob + i * 16 + r + 8 * hl;
        const float val = acc[i][r] + bias[o];
        const size_t idx = (size_t)b * DMODEL * LL + (size_t)o * LL + (l0 + q15);
        if constexpr (OUTMODE == 2) ((float*)Yv)[idx] = val;
        else                        ((__bf16*)Yv)[idx] = (__bf16)val;
      }
  }
}

// ============================================================
// Banded local attention per (b, segment, head).
// qT/kT are (b, L, 512) transposed; vh is (b, 512, L); ohT out (b, L, 512).
// __launch_bounds__(256, 2): ~512 VGPRs/wave available -> no acc spills.
// ============================================================
__global__ __launch_bounds__(256, 2) void
attn_kernel(const __bf16* __restrict__ qT,
            const __bf16* __restrict__ kT,
            const __bf16* __restrict__ vh,
            const int* __restrict__ mask,
            __bf16* __restrict__ ohT)
{
  __shared__ __bf16 sKt[WIN * DH];             // 24 KB, (w, d) pitch 64
  __shared__ __bf16 sV [DH * WIN];             // 24 KB, (d, w) pitch 192
  __shared__ __bf16 chnk[8 * 16 * 40];         // per-wave P^T strip

  const int lane = threadIdx.x & 31;
  const int wave = threadIdx.x >> 5;
  const int hl   = lane >> 4;
  const int q15  = lane & 15;

  int idx = blockIdx.x;
  const int h  = idx & 7;  idx >>= 3;
  const int ns = idx & 31; idx >>= 5;
  const int b  = idx;
  const int   pos0  = ns * LSEG - W2;
  const int   lrow0 = wave * 16;
  const int   lg    = ns * LSEG + lrow0 + q15;
  const bool  interior = (ns > 0) && (ns < NSEG - 1);
  const size_t tbase = (size_t)b * LL * DMODEL;
  const size_t vbase = (size_t)b * DMODEL * LL + (size_t)h * DH * LL;

  // ---- stage K (transposed) and V windows into LDS ----
  if (!interior) {
    for (int i = threadIdx.x; i < WIN * (DH / 2); i += 256) {   // K: 32 dwords/row
      const int w = i >> 5, c = i & 31;
      const int pos = pos0 + w;
      unsigned val = 0;
      if (pos >= 0 && pos < LL)
        val = *(const unsigned*)(kT + tbase + (size_t)pos * DMODEL + h * DH + c * 2);
      ((unsigned*)sKt)[i] = val;
    }
    for (int i = threadIdx.x; i < DH * (WIN / 2); i += 256) {   // V: 96 dwords/row
      const int row = i / (WIN / 2);
      const int pos = pos0 + (i % (WIN / 2)) * 2;
      unsigned val = 0;
      if (pos >= 0 && pos < LL)
        val = *(const unsigned*)(vh + vbase + (size_t)row * LL + pos);
      ((unsigned*)sV)[i] = val;
    }
  }
  if (wave == 0 && interior) {
    tdm_load_2d((unsigned)(size_t)(void*)sKt,
                kT + tbase + (size_t)pos0 * DMODEL + h * DH, WIN, DH, DMODEL);
    tdm_load_2d((unsigned)(size_t)(void*)sV,
                vh + vbase + pos0, DH, WIN, LL);
    __builtin_amdgcn_s_wait_tensorcnt(0);
  }
  __syncthreads();

  // key/pad validity per w-tile t (w = t*16 + q15)
  unsigned kvmask = 0;
  #pragma unroll
  for (int t = 0; t < 12; ++t) {
    const int pos = pos0 + t * 16 + q15;
    const bool ok = (pos >= 0) && (pos < LL) && (mask[b * LL + pos] != 0);
    kvmask |= (ok ? 1u : 0u) << t;
  }

  // ---- S = Q^T K : 12 accumulator tiles, K-dim 64, double-buffered B ----
  v8f acc[12];
  #pragma unroll
  for (int t = 0; t < 12; ++t) acc[t] = (v8f){};

  #pragma unroll
  for (int d0 = 0; d0 < DH; d0 += 32) {
    const __bf16* qr = qT + tbase + (size_t)lg * DMODEL + h * DH + d0 + hl * 16;
    const v16bf a = ld16pair(qr, qr + 8);
    const __bf16* kbase = sKt + q15 * DH + d0 + hl * 16;
    v16bf bb = ld16pair(kbase, kbase + 8);           // t = 0
    #pragma unroll
    for (int t = 0; t < 12; ++t) {
      const v16bf cur = bb;
      if (t < 11) {                                  // prefetch next B-frag
        const __bf16* kp = kbase + (t + 1) * 16 * DH;
        bb = ld16pair(kp, kp + 8);
      }
      acc[t] = __builtin_amdgcn_wmma_f32_16x16x32_bf16(false, a, false, cur,
                                                       (short)0, acc[t], false, false);
    }
  }

  // ---- scale + band mask + row softmax (row = 16 lanes of one VGPR) ----
  const float scale = 0.125f;
  float inv8[8];
  #pragma unroll
  for (int r = 0; r < 8; ++r) {
    const int l = lrow0 + r + 8 * hl;
    float mx = -3.0e38f;
    #pragma unroll
    for (int t = 0; t < 12; ++t) {
      const int w = t * 16 + q15;
      const bool keep = ((kvmask >> t) & 1u) && (w >= l) && (w <= l + 64);
      const float s = keep ? acc[t][r] * scale : -1.0e9f;
      acc[t][r] = s;
      mx = fmaxf(mx, s);
    }
    #pragma unroll
    for (int m = 1; m <= 8; m <<= 1) mx = fmaxf(mx, __shfl_xor(mx, m, 32));
    float sum = 0.f;
    #pragma unroll
    for (int t = 0; t < 12; ++t) {
      const float e = __expf(acc[t][r] - mx);
      acc[t][r] = e;
      sum += e;
    }
    #pragma unroll
    for (int m = 1; m <= 8; m <<= 1) sum += __shfl_xor(sum, m, 32);
    inv8[r] = __builtin_amdgcn_rcpf(sum);
  }

  // ---- O = V * P^T : 4 d-tiles, K=192 in 6 steps; P transposed via
  //      wave-private LDS strip; A-fragments double-buffered ----
  v8f oacc[4];
  #pragma unroll
  for (int mt = 0; mt < 4; ++mt) oacc[mt] = (v8f){};
  __bf16* myc = chnk + wave * 16 * 40;

  #pragma unroll
  for (int kk = 0; kk < 6; ++kk) {
    #pragma unroll
    for (int tt = 0; tt < 2; ++tt) {
      const int t = kk * 2 + tt;
      #pragma unroll
      for (int r = 0; r < 8; ++r)
        myc[(r + 8 * hl) * 40 + tt * 16 + q15] = (__bf16)(acc[t][r] * inv8[r]);
    }
    const __bf16* cp = myc + q15 * 40 + hl * 16;
    const v16bf bb = ld16pair(cp, cp + 8);
    const __bf16* vbase0 = sV + q15 * WIN + kk * 32 + hl * 16;
    v16bf va = ld16pair(vbase0, vbase0 + 8);         // mt = 0
    #pragma unroll
    for (int mt = 0; mt < 4; ++mt) {
      const v16bf cur = va;
      if (mt < 3) {                                  // prefetch next A-frag
        const __bf16* vp = vbase0 + (mt + 1) * 16 * WIN;
        va = ld16pair(vp, vp + 8);
      }
      oacc[mt] = __builtin_amdgcn_wmma_f32_16x16x32_bf16(false, cur, false, bb,
                                                         (short)0, oacc[mt], false, false);
    }
  }

  // store o_hat transposed (b, L, 512): per-lane contiguous 16B stores
  __bf16* orow = ohT + tbase + (size_t)lg * DMODEL + h * DH;
  #pragma unroll
  for (int mt = 0; mt < 4; ++mt) {
    v8bf ov;
    #pragma unroll
    for (int r = 0; r < 8; ++r) ov[r] = (__bf16)oacc[mt][r];
    *(v8bf*)(orow + mt * 16 + 8 * hl) = ov;
  }
}

// ============================================================
extern "C" void kernel_launch(void* const* d_in, const int* in_sizes, int n_in,
                              void* d_out, int out_size, void* d_ws, size_t ws_size,
                              hipStream_t stream) {
  (void)in_sizes; (void)n_in; (void)out_size; (void)ws_size;
  const float* q    = (const float*)d_in[0];
  const float* k    = (const float*)d_in[1];
  const float* v    = (const float*)d_in[2];
  const int*   mask = (const int*)d_in[3];
  const float* Wq   = (const float*)d_in[4];
  const float* Wqb  = (const float*)d_in[5];
  const float* Wk   = (const float*)d_in[6];
  const float* Wkb  = (const float*)d_in[7];
  const float* Wv   = (const float*)d_in[8];
  const float* Wvb  = (const float*)d_in[9];
  const float* Wo   = (const float*)d_in[10];
  const float* Wob  = (const float*)d_in[11];

  const size_t S = (size_t)BB * DMODEL * LL;   // elements per tensor
  __bf16* qT  = (__bf16*)d_ws;                 // (b, L, 512)
  __bf16* kT  = qT + S;                        // (b, L, 512)
  __bf16* vh  = kT + S;                        // (b, 512, L)
  __bf16* ohT = vh + S;                        // (b, L, 512)

  const dim3 blk(256);
  const dim3 gproj(LL / 16, BB);
  proj_kernel<false, 1><<<gproj, blk, 0, stream>>>(Wq, Wqb, q, qT);
  proj_kernel<false, 1><<<gproj, blk, 0, stream>>>(Wk, Wkb, k, kT);
  proj_kernel<false, 0><<<gproj, blk, 0, stream>>>(Wv, Wvb, v, vh);
  attn_kernel<<<dim3(BB * NSEG * HN), blk, 0, stream>>>(qT, kT, vh, mask, ohT);
  proj_kernel<true, 2><<<gproj, blk, 0, stream>>>(Wo, Wob, ohT, (float*)d_out);
}
